// SPAIRGlimpseMaskDecoder_15470472200211
// MI455X (gfx1250) — matrix operand order repaired and checked by reading, not compiled, not run on hardware
//
#include <hip/hip_runtime.h>
#include <hip/hip_bf16.h>
#include <math.h>

// ---------------------------------------------------------------------------
// Sizes from the reference
// ---------------------------------------------------------------------------
#define N4 32768
#define N3 65536
#define N2 131072
#define N1 262144
#define KNB 16           // neighbors per output point (edges are contiguous per dst)

typedef __attribute__((ext_vector_type(16))) _Float16 v16h;
typedef __attribute__((ext_vector_type(8)))  _Float16 v8h;
typedef __attribute__((ext_vector_type(8)))  float    v8f;

#define DEVINL __device__ __forceinline__

DEVINL v8f wmma_f16(v16h a, v16h b, v8f c) {
  // D = A(16x32 f16) * B(32x16 f16) + C(16x16 f32)
  return __builtin_amdgcn_wmma_f32_16x16x32_f16(
      /*neg_a=*/false, a, /*neg_b=*/false, b,
      /*c_mod=*/(short)0, c, /*reuse_a=*/false, /*reuse_b=*/false);
}

// ---------------------------------------------------------------------------
// Async global->LDS copy (GLOBAL_LOAD_ASYNC_TO_LDS_B128, ASYNCcnt).
// IOFFSET applies to BOTH the global and the LDS address, so one base pair
// plus literal offsets covers a contiguous per-lane copy region.
// LDS byte address = low 32 bits of the generic pointer (LDS aperture keeps
// the offset in addr[31:0]).
// ---------------------------------------------------------------------------
DEVINL unsigned ldsoff(const void* p) { return (unsigned)(uintptr_t)p; }

#define ASYNC_B128(ldsaddr, gaddr, imm)                                        \
  asm volatile("global_load_async_to_lds_b128 %0, %1, off offset:" #imm        \
               :: "v"(ldsaddr), "v"(gaddr) : "memory")

DEVINL void wait_async0() { asm volatile("s_wait_asynccnt 0x0" ::: "memory"); }

// A fragment (16-bit A, 16x32 MxK): lane = m + 16*h.
//   elems 0..7  <- K = 32*kt + 8*h + (0..7)
//   elems 8..15 <- K = 32*kt + 16 + 8*h + (0..7)
DEVINL v16h loadA(const _Float16* Abase, int kpad, int lane, int kt) {
  int m = lane & 15, h = lane >> 4;
  const _Float16* row = Abase + m * kpad + 32 * kt + 8 * h;
  v8h s0 = *reinterpret_cast<const v8h*>(row);
  v8h s1 = *reinterpret_cast<const v8h*>(row + 16);
  return __builtin_shufflevector(s0, s1, 0,1,2,3,4,5,6,7,8,9,10,11,12,13,14,15);
}

// B fragment (16-bit B, 32x16 KxN) from W^T stored [N][KPAD] in LDS:
//   lane = n + 16*h holds column n, elems 0..7 <- K = 32*kt + 8*h + (0..7),
//   elems 8..15 <- K = 32*kt + 16 + 8*h + (0..7)
DEVINL v16h loadB(const _Float16* WT, int kpad, int lane, int nt, int kt) {
  int n = lane & 15, h = lane >> 4;
  const _Float16* col = WT + (16 * nt + n) * kpad + 32 * kt + 8 * h;
  v8h s0 = *reinterpret_cast<const v8h*>(col);
  v8h s1 = *reinterpret_cast<const v8h*>(col + 16);
  return __builtin_shufflevector(s0, s1, 0,1,2,3,4,5,6,7,8,9,10,11,12,13,14,15);
}

// Branch-free CELU(alpha=1) using the hardware transcendental (v_exp_f32).
// exp(0) == 1 exactly, so the second term is exactly 0 for x >= 0.
DEVINL float celu1(float x) {
  return fmaxf(x, 0.f) + (__expf(fminf(x, 0.f)) - 1.f);
}
// log_sigmoid, stable, branch-free, hardware exp/log.
DEVINL float logsig(float x) {
  return fminf(x, 0.f) - __logf(1.f + __expf(-fabsf(x)));
}

// ---------------------------------------------------------------------------
// z_mask f32 -> f16 pre-pass (halves the random-gather bytes in level 3)
// ---------------------------------------------------------------------------
struct h4 { _Float16 x, y, z, w; };

__global__ void zcast_kernel(const float4* __restrict__ in, h4* __restrict__ out, int n4) {
  int i = blockIdx.x * blockDim.x + threadIdx.x;
  if (i < n4) {
    float4 v = in[i];
    h4 o; o.x = (_Float16)v.x; o.y = (_Float16)v.y; o.z = (_Float16)v.z; o.w = (_Float16)v.w;
    out[i] = o;
  }
}

// ---------------------------------------------------------------------------
// Level 3: c_in=64(+3 -> Kpad 96), c_mid=64 (4 N-tiles), c_out=32 (2 N-tiles)
// One wave per output point; 8 waves (points) per block.
// ---------------------------------------------------------------------------
__global__ __launch_bounds__(256) void pc3_kernel(
    const _Float16* __restrict__ z16, const float* __restrict__ pos4,
    const float* __restrict__ pos3,
    const float* __restrict__ w1a, const float* __restrict__ b1a,
    const float* __restrict__ w1b, const float* __restrict__ b1b,
    const int* __restrict__ src3, _Float16* __restrict__ x3) {
  __shared__ alignas(16) _Float16 w1aT[64 * 96];     // W1a^T [N=64][Kpad=96]
  __shared__ alignas(16) _Float16 w1bT[32 * 64];     // W1b^T [N=32][Kpad=64]
  __shared__ alignas(16) _Float16 Abuf[8][16 * 96];  // per-wave edge features
  __shared__ alignas(16) _Float16 Hbuf[8][16 * 64];  // per-wave hidden (celu'd)

  const int tid = threadIdx.x;
  for (int i = tid; i < 64 * 96; i += 256) {
    int n = i / 96, k = i % 96;
    w1aT[i] = (k < 67) ? (_Float16)w1a[k * 64 + n] : (_Float16)0.f;
  }
  for (int i = tid; i < 32 * 64; i += 256) {
    int n = i / 64, k = i % 64;
    w1bT[i] = (_Float16)w1b[k * 32 + n];
  }
  __syncthreads();

  const int wave = tid >> 5, lane = tid & 31;
  const int m = lane & 15, h = lane >> 4;
  const int p = blockIdx.x * 8 + wave;

  _Float16* A = Abuf[wave];
  _Float16* H = Hbuf[wave];
  _Float16* arow = A + m * 96;
  const v8h z8 = {};

  const int src = src3[p * KNB + m];
  // two lanes per edge: lane half h copies channels [32h, 32h+32) = 64 B,
  // contiguous in global AND in LDS -> 4 async b128 copies.
  {
    unsigned la = ldsoff(arow + 32 * h);
    unsigned long long ga = (unsigned long long)(uintptr_t)(z16 + (size_t)src * 64 + 32 * h);
    ASYNC_B128(la, ga, 0);
    ASYNC_B128(la, ga, 16);
    ASYNC_B128(la, ga, 32);
    ASYNC_B128(la, ga, 48);
  }
  // zero K-padding [64,96) with wide stores, then overwrite rel (in-order LDS)
  *reinterpret_cast<v8h*>(arow + 64 + 16 * h) = z8;
  *reinterpret_cast<v8h*>(arow + 72 + 16 * h) = z8;
  if (h == 0) {
    arow[64] = (_Float16)(pos4[src * 3 + 0] - pos3[p * 3 + 0]);
    arow[65] = (_Float16)(pos4[src * 3 + 1] - pos3[p * 3 + 1]);
    arow[66] = (_Float16)(pos4[src * 3 + 2] - pos3[p * 3 + 2]);
  }
  wait_async0();  // async LDS writes visible before fragment reads

  v16h a0 = loadA(A, 96, lane, 0);
  v16h a1 = loadA(A, 96, lane, 1);
  v16h a2 = loadA(A, 96, lane, 2);

#pragma unroll
  for (int nt = 0; nt < 4; ++nt) {
    float bias = b1a[nt * 16 + m];
    v8f acc;
#pragma unroll
    for (int r = 0; r < 8; ++r) acc[r] = bias;
    acc = wmma_f16(a0, loadB(w1aT, 96, lane, nt, 0), acc);
    acc = wmma_f16(a1, loadB(w1aT, 96, lane, nt, 1), acc);
    acc = wmma_f16(a2, loadB(w1aT, 96, lane, nt, 2), acc);
#pragma unroll
    for (int r = 0; r < 8; ++r)  // D: row r+8h, col m
      H[(r + 8 * h) * 64 + nt * 16 + m] = (_Float16)celu1(acc[r]);
  }

  v16h h0 = loadA(H, 64, lane, 0);
  v16h h1 = loadA(H, 64, lane, 1);
#pragma unroll
  for (int nt = 0; nt < 2; ++nt) {
    float bias = b1b[nt * 16 + m];
    v8f acc;
#pragma unroll
    for (int r = 0; r < 8; ++r) acc[r] = bias;
    acc = wmma_f16(h0, loadB(w1bT, 64, lane, nt, 0), acc);
    acc = wmma_f16(h1, loadB(w1bT, 64, lane, nt, 1), acc);
    float mx = acc[0];
#pragma unroll
    for (int r = 1; r < 8; ++r) mx = fmaxf(mx, acc[r]);
    mx = fmaxf(mx, __shfl_xor(mx, 16, 32));  // merge rows 0-7 with 8-15
    float o = celu1(mx);
    if (h == 0) x3[(size_t)p * 32 + nt * 16 + m] = (_Float16)o;
  }
}

// ---------------------------------------------------------------------------
// Level 2: c_in=32(+3 -> Kpad 64), c_mid=16, c_out=16
// ---------------------------------------------------------------------------
__global__ __launch_bounds__(256) void pc2_kernel(
    const _Float16* __restrict__ x3, const float* __restrict__ pos3,
    const float* __restrict__ pos2,
    const float* __restrict__ w2a, const float* __restrict__ b2a,
    const float* __restrict__ w2b, const float* __restrict__ b2b,
    const int* __restrict__ src2, _Float16* __restrict__ x2) {
  __shared__ alignas(16) _Float16 w2aT[16 * 64];
  __shared__ alignas(16) _Float16 w2bT[16 * 32];
  __shared__ alignas(16) _Float16 Abuf[8][16 * 64];
  __shared__ alignas(16) _Float16 Hbuf[8][16 * 32];

  const int tid = threadIdx.x;
  for (int i = tid; i < 16 * 64; i += 256) {
    int n = i / 64, k = i % 64;
    w2aT[i] = (k < 35) ? (_Float16)w2a[k * 16 + n] : (_Float16)0.f;
  }
  for (int i = tid; i < 16 * 32; i += 256) {
    int n = i / 32, k = i % 32;
    w2bT[i] = (k < 16) ? (_Float16)w2b[k * 16 + n] : (_Float16)0.f;
  }
  __syncthreads();

  const int wave = tid >> 5, lane = tid & 31;
  const int m = lane & 15, h = lane >> 4;
  const int p = blockIdx.x * 8 + wave;

  _Float16* A = Abuf[wave];
  _Float16* H = Hbuf[wave];
  _Float16* arow = A + m * 64;
  const v8h z8 = {};

  const int src = src2[p * KNB + m];
  // lane half h copies channels [16h, 16h+16) = 32 B -> 2 async b128 copies
  {
    unsigned la = ldsoff(arow + 16 * h);
    unsigned long long ga = (unsigned long long)(uintptr_t)(x3 + (size_t)src * 32 + 16 * h);
    ASYNC_B128(la, ga, 0);
    ASYNC_B128(la, ga, 16);
  }
  // zero K-padding [32,64), then rel overwrite
  *reinterpret_cast<v8h*>(arow + 32 + 16 * h) = z8;
  *reinterpret_cast<v8h*>(arow + 40 + 16 * h) = z8;
  if (h == 0) {
    arow[32] = (_Float16)(pos3[src * 3 + 0] - pos2[p * 3 + 0]);
    arow[33] = (_Float16)(pos3[src * 3 + 1] - pos2[p * 3 + 1]);
    arow[34] = (_Float16)(pos3[src * 3 + 2] - pos2[p * 3 + 2]);
  }
  // zero H K-padding cols [16,32): lane (m,h) zeros H[m][16+8h .. +7]
  *reinterpret_cast<v8h*>(H + m * 32 + 16 + 8 * h) = z8;
  wait_async0();

  v16h a0 = loadA(A, 64, lane, 0);
  v16h a1 = loadA(A, 64, lane, 1);

  float bias = b2a[m];
  v8f acc;
#pragma unroll
  for (int r = 0; r < 8; ++r) acc[r] = bias;
  acc = wmma_f16(a0, loadB(w2aT, 64, lane, 0, 0), acc);
  acc = wmma_f16(a1, loadB(w2aT, 64, lane, 0, 1), acc);
#pragma unroll
  for (int r = 0; r < 8; ++r)
    H[(r + 8 * h) * 32 + m] = (_Float16)celu1(acc[r]);

  v16h hf = loadA(H, 32, lane, 0);
  float bias2 = b2b[m];
  v8f acc2;
#pragma unroll
  for (int r = 0; r < 8; ++r) acc2[r] = bias2;
  acc2 = wmma_f16(hf, loadB(w2bT, 32, lane, 0, 0), acc2);
  float mx = acc2[0];
#pragma unroll
  for (int r = 1; r < 8; ++r) mx = fmaxf(mx, acc2[r]);
  mx = fmaxf(mx, __shfl_xor(mx, 16, 32));
  float o = celu1(mx);
  if (h == 0) x2[(size_t)p * 16 + m] = (_Float16)o;
}

// ---------------------------------------------------------------------------
// Level 1: c_in=16(+3 -> Kpad 32), c_mid=8 (N padded to 16), c_out=8,
// then logits = out @ wlin + blin, log_sigmoid.
// ---------------------------------------------------------------------------
__global__ __launch_bounds__(256) void pc1_kernel(
    const _Float16* __restrict__ x2, const float* __restrict__ pos2,
    const float* __restrict__ pos1,
    const float* __restrict__ w3a, const float* __restrict__ b3a,
    const float* __restrict__ w3b, const float* __restrict__ b3b,
    const float* __restrict__ wlin, const float* __restrict__ blin,
    const int* __restrict__ src1, float* __restrict__ out) {
  __shared__ alignas(16) _Float16 w3aT[16 * 32];
  __shared__ alignas(16) _Float16 w3bT[16 * 32];
  __shared__ alignas(16) _Float16 Abuf[8][16 * 32];
  __shared__ alignas(16) _Float16 Hbuf[8][16 * 32];

  const int tid = threadIdx.x;
  for (int i = tid; i < 16 * 32; i += 256) {
    int n = i / 32, k = i % 32;
    w3aT[i] = (n < 8 && k < 19) ? (_Float16)w3a[k * 8 + n] : (_Float16)0.f;
    w3bT[i] = (n < 8 && k < 8)  ? (_Float16)w3b[k * 8 + n] : (_Float16)0.f;
  }
  __syncthreads();

  const int wave = tid >> 5, lane = tid & 31;
  const int m = lane & 15, h = lane >> 4;
  const int p = blockIdx.x * 8 + wave;

  _Float16* A = Abuf[wave];
  _Float16* H = Hbuf[wave];
  _Float16* arow = A + m * 32;
  const v8h z8 = {};

  const int src = src1[p * KNB + m];
  // lane half h copies channels [8h, 8h+8) = 16 B -> 1 async b128 copy
  {
    unsigned la = ldsoff(arow + 8 * h);
    unsigned long long ga = (unsigned long long)(uintptr_t)(x2 + (size_t)src * 16 + 8 * h);
    ASYNC_B128(la, ga, 0);
  }
  // zero K-padding [16,32), then rel overwrite
  *reinterpret_cast<v8h*>(arow + 16 + 8 * h) = z8;
  if (h == 0) {
    arow[16] = (_Float16)(pos2[src * 3 + 0] - pos1[p * 3 + 0]);
    arow[17] = (_Float16)(pos2[src * 3 + 1] - pos1[p * 3 + 1]);
    arow[18] = (_Float16)(pos2[src * 3 + 2] - pos1[p * 3 + 2]);
  }
  // zero H K-padding cols [16,32)
  *reinterpret_cast<v8h*>(H + m * 32 + 16 + 8 * h) = z8;
  wait_async0();

  v16h a0 = loadA(A, 32, lane, 0);
  float bias = (m < 8) ? b3a[m] : 0.f;
  v8f acc;
#pragma unroll
  for (int r = 0; r < 8; ++r) acc[r] = bias;
  acc = wmma_f16(a0, loadB(w3aT, 32, lane, 0, 0), acc);
#pragma unroll
  for (int r = 0; r < 8; ++r)
    H[(r + 8 * h) * 32 + m] = (_Float16)celu1(acc[r]);  // cols 8..15 are exactly 0

  v16h hf = loadA(H, 32, lane, 0);
  float bias2 = (m < 8) ? b3b[m] : 0.f;
  v8f acc2;
#pragma unroll
  for (int r = 0; r < 8; ++r) acc2[r] = bias2;
  acc2 = wmma_f16(hf, loadB(w3bT, 32, lane, 0, 0), acc2);
  float mx = acc2[0];
#pragma unroll
  for (int r = 1; r < 8; ++r) mx = fmaxf(mx, acc2[r]);
  mx = fmaxf(mx, __shfl_xor(mx, 16, 32));
  float o = celu1(mx);

  // logits = out(8) . wlin + blin ; both 16-lane halves hold identical values
  float val = (m < 8) ? o * wlin[m] : 0.f;
  val += __shfl_xor(val, 8, 32);
  val += __shfl_xor(val, 4, 32);
  val += __shfl_xor(val, 2, 32);
  val += __shfl_xor(val, 1, 32);
  if (lane == 0) out[p] = logsig(val + blin[0]);
}

// ---------------------------------------------------------------------------
// Launch
// ---------------------------------------------------------------------------
extern "C" void kernel_launch(void* const* d_in, const int* in_sizes, int n_in,
                              void* d_out, int out_size, void* d_ws, size_t ws_size,
                              hipStream_t stream) {
  const float* z_mask = (const float*)d_in[0];
  const float* pos4   = (const float*)d_in[1];
  const float* pos3   = (const float*)d_in[2];
  const float* pos2   = (const float*)d_in[3];
  const float* pos1   = (const float*)d_in[4];
  // d_in[5..8] = g4..g1 (unused by the reference computation)
  const float* w1a = (const float*)d_in[9];
  const float* b1a = (const float*)d_in[10];
  const float* w1b = (const float*)d_in[11];
  const float* b1b = (const float*)d_in[12];
  const float* w2a = (const float*)d_in[13];
  const float* b2a = (const float*)d_in[14];
  const float* w2b = (const float*)d_in[15];
  const float* b2b = (const float*)d_in[16];
  const float* w3a = (const float*)d_in[17];
  const float* b3a = (const float*)d_in[18];
  const float* w3b = (const float*)d_in[19];
  const float* b3b = (const float*)d_in[20];
  const float* wlin = (const float*)d_in[21];
  const float* blin = (const float*)d_in[22];
  const int* src3 = (const int*)d_in[23];
  const int* src2 = (const int*)d_in[25];
  const int* src1 = (const int*)d_in[27];
  float* out = (float*)d_out;

  // workspace: f16 activations + f16 z_mask copy (12 MB total)
  _Float16* ws  = (_Float16*)d_ws;
  _Float16* x3  = ws;                                        // N3*32 halves
  _Float16* x2  = x3 + (size_t)N3 * 32;                      // N2*16 halves
  _Float16* z16 = x2 + (size_t)N2 * 16;                      // N4*64 halves

  int nz4 = (N4 * 64) / 4;
  zcast_kernel<<<(nz4 + 255) / 256, 256, 0, stream>>>(
      (const float4*)z_mask, (h4*)z16, nz4);

  pc3_kernel<<<N3 / 8, 256, 0, stream>>>(z16, pos4, pos3, w1a, b1a, w1b, b1b, src3, x3);
  pc2_kernel<<<N2 / 8, 256, 0, stream>>>(x3, pos3, pos2, w2a, b2a, w2b, b2b, src2, x2);
  pc1_kernel<<<N1 / 8, 256, 0, stream>>>(x2, pos2, pos1, w3a, b3a, w3b, b3b,
                                         wlin, blin, src1, out);
}
